// GMGN_76510547411308
// MI455X (gfx1250) — matrix operand order repaired and checked by reading, not compiled, not run on hardware
//
#include <hip/hip_runtime.h>

typedef __attribute__((ext_vector_type(16))) _Float16 v16h;
typedef __attribute__((ext_vector_type(8)))  float    v8f;

#define HID   2048
#define MROWS 16425
#define MPAD  16512   // 129 * 128
#define BM    128
#define BN    128
#define BK    32      // one wmma k-step (16x16x32 f16)

// ---------------------------------------------------------------------------
// Weight convert+transpose: w (f32, [k][n] row-major KxN) -> wt (f16, [n][k])
// ---------------------------------------------------------------------------
__global__ __launch_bounds__(256) void wt_convert(const float* __restrict__ w,
                                                  _Float16* __restrict__ wt) {
  int idx = blockIdx.x * 256 + threadIdx.x;       // over HID*HID, output-coalesced
  int n = idx >> 11;                              // idx / HID
  int k = idx & (HID - 1);
  wt[idx] = (_Float16)w[k * HID + n];
}

// ---------------------------------------------------------------------------
// Layer 0: act[m][n] = relu(xs[m] * w0[n] + b0[n]) as f16, padded rows use x=0
// ---------------------------------------------------------------------------
__global__ __launch_bounds__(256) void layer0(const float* __restrict__ xs,
                                              const float* __restrict__ w0,
                                              const float* __restrict__ b0,
                                              _Float16* __restrict__ act) {
  int idx = blockIdx.x * 256 + threadIdx.x;       // over MPAD*HID
  int m = idx >> 11;
  int n = idx & (HID - 1);
  float x = (m < MROWS) ? xs[m] : 0.0f;
  float v = fmaxf(fmaf(x, w0[n], b0[n]), 0.0f);
  act[idx] = (_Float16)v;
}

// ---------------------------------------------------------------------------
// WMMA GEMM: C = relu(A @ W + b), A: MPADxHID f16, Wt: HID(n)xHID(k) f16
// block = 256 threads = 8 waves; tile 128x128; wave subtile 32x64
// Double-buffered LDS: one barrier per K-step, global loads overlap WMMA.
// ---------------------------------------------------------------------------
__device__ __forceinline__ v16h ld_frag(const _Float16* lo, const _Float16* hi) {
  union { v16h v; uint4 u[2]; } r;
  r.u[0] = *(const uint4*)lo;   // elements 0..7
  r.u[1] = *(const uint4*)hi;   // elements 8..15
  return r.v;
}

__global__ __launch_bounds__(256) void gemm_relu(const _Float16* __restrict__ A,
                                                 const _Float16* __restrict__ Wt,
                                                 const float* __restrict__ bias,
                                                 _Float16* __restrict__ C) {
  __shared__ _Float16 lA[2][BM * BK];   // [m][k], double-buffered (2 x 8KB)
  __shared__ _Float16 lB[2][BN * BK];   // [n][k], double-buffered (2 x 8KB)

  const int tid   = threadIdx.x;
  const int lane  = tid & 31;
  const int wave  = tid >> 5;
  const int m0    = blockIdx.x * BM;
  const int n0    = blockIdx.y * BN;
  const int wm    = (wave & 3) * 32;   // wave row offset in tile
  const int wn    = (wave >> 2) * 64;  // wave col offset in tile
  const int lhalf = lane >> 4;         // 0/1
  const int lmod  = lane & 15;

  // staging chunks: 512 x 16B per tile, 2 per thread
  const int c0  = tid;                 // chunk 0
  const int c1  = tid + 256;           // chunk 1
  const int r0  = c0 >> 2, kc0 = (c0 & 3) << 3;
  const int r1  = c1 >> 2, kc1 = (c1 & 3) << 3;
  const int s0  = r0 * BK + kc0;       // LDS half-offset for chunk 0
  const int s1  = r1 * BK + kc1;       // LDS half-offset for chunk 1

  const _Float16* gA0 = A  + (size_t)(m0 + r0) * HID + kc0;
  const _Float16* gA1 = A  + (size_t)(m0 + r1) * HID + kc1;
  const _Float16* gB0 = Wt + (size_t)(n0 + r0) * HID + kc0;
  const _Float16* gB1 = Wt + (size_t)(n0 + r1) * HID + kc1;

  // fragment LDS half-offsets (ISA 7.12.2 16-bit layouts)
  int aoff[2], boff[4];
#pragma unroll
  for (int i = 0; i < 2; ++i) aoff[i] = (wm + i * 16 + lmod) * BK + lhalf * 8;
#pragma unroll
  for (int j = 0; j < 4; ++j) boff[j] = (wn + j * 16 + lmod) * BK + lhalf * 8;

  v8f acc[2][4] = {};

  // ---- prologue: stage K-tile 0 into buffer 0 ----
  uint4 ra0 = *(const uint4*)gA0, ra1 = *(const uint4*)gA1;
  uint4 rb0 = *(const uint4*)gB0, rb1 = *(const uint4*)gB1;
  gA0 += BK; gA1 += BK; gB0 += BK; gB1 += BK;
  *(uint4*)&lA[0][s0] = ra0;
  *(uint4*)&lA[0][s1] = ra1;
  *(uint4*)&lB[0][s0] = rb0;
  *(uint4*)&lB[0][s1] = rb1;
  __syncthreads();

  int buf = 0;
  for (int k0 = 0; k0 < HID; k0 += BK, buf ^= 1) {
    const bool more = (k0 + BK) < HID;
    if (more) {
      // issue next-tile global loads early; latency hidden by WMMA below
      ra0 = *(const uint4*)gA0;
      ra1 = *(const uint4*)gA1;
      rb0 = *(const uint4*)gB0;
      rb1 = *(const uint4*)gB1;
      gA0 += BK; gA1 += BK; gB0 += BK; gB1 += BK;
      // warm caches for the tile after next (global_prefetch_b8, WGP scope)
      __builtin_prefetch(gA0, 0, 3);
      __builtin_prefetch(gB0, 0, 3);
    }

    // ---- fragments from current buffer ----
    v16h af[2];
#pragma unroll
    for (int i = 0; i < 2; ++i) {
      const _Float16* base = &lA[buf][aoff[i]];
      af[i] = ld_frag(base, base + 16);  // K = {lhalf*8+0..7, 16+lhalf*8+0..7}
    }
    v16h bf[4];
#pragma unroll
    for (int j = 0; j < 4; ++j) {
      const _Float16* base = &lB[buf][boff[j]];
      bf[j] = ld_frag(base, base + 16);  // K = {lhalf*8+0..7, 16+lhalf*8+0..7}
    }

#pragma unroll
    for (int i = 0; i < 2; ++i)
#pragma unroll
      for (int j = 0; j < 4; ++j)
        acc[i][j] = __builtin_amdgcn_wmma_f32_16x16x32_f16(
            false, af[i], false, bf[j], (short)0, acc[i][j], false, false);

    if (more) {
      // write next tile into the alternate buffer; prior readers of it are
      // guaranteed done by the previous end-of-iteration barrier.
      const int nb = buf ^ 1;
      *(uint4*)&lA[nb][s0] = ra0;
      *(uint4*)&lA[nb][s1] = ra1;
      *(uint4*)&lB[nb][s0] = rb0;
      *(uint4*)&lB[nb][s1] = rb1;
      __syncthreads();
    }
  }

  // ---- epilogue: bias + relu + f16 store ----
#pragma unroll
  for (int i = 0; i < 2; ++i) {
#pragma unroll
    for (int j = 0; j < 4; ++j) {
      int ncol = n0 + wn + j * 16 + lmod;
      float bv = bias[ncol];
#pragma unroll
      for (int r = 0; r < 8; ++r) {
        int mrow = m0 + wm + i * 16 + lhalf * 8 + r;
        float v = fmaxf(acc[i][j][r] + bv, 0.0f);
        C[(size_t)mrow * HID + ncol] = (_Float16)v;
      }
    }
  }
}

// ---------------------------------------------------------------------------
// Final layer: amg[m] = relu(dot(act[m], w4) + b4), one wave per row
// ---------------------------------------------------------------------------
__global__ __launch_bounds__(256) void final_dot(const _Float16* __restrict__ act,
                                                 const float* __restrict__ w4,
                                                 const float* __restrict__ b4,
                                                 float* __restrict__ amg) {
  int row  = blockIdx.x * 8 + (threadIdx.x >> 5);
  int lane = threadIdx.x & 31;
  if (row >= MROWS) return;
  const _Float16* a = act + (size_t)row * HID;
  float s = 0.0f;
  for (int k = lane; k < HID; k += 32)
    s = fmaf((float)a[k], w4[k], s);
#pragma unroll
  for (int off = 16; off > 0; off >>= 1)
    s += __shfl_xor(s, off, 32);
  if (lane == 0) amg[row] = fmaxf(s + b4[0], 0.0f);
}

// ---------------------------------------------------------------------------
// Multigrid prolongation: out[2i]=in[i], out[2i+1]=avg; then 8-pt override
// ---------------------------------------------------------------------------
__global__ __launch_bounds__(256) void interp_scatter(const float* __restrict__ in,
                                                      int n_in,
                                                      float* __restrict__ out,
                                                      const int* __restrict__ nbr_row,
                                                      const float* __restrict__ band) {
  int j = blockIdx.x * 256 + threadIdx.x;
  int n_out = 2 * n_in - 1;
  if (j >= n_out) return;
  float v;
  int h = j >> 1;
  if ((j & 1) == 0) v = in[h];
  else              v = 0.5f * (in[h] + in[h + 1]);
#pragma unroll
  for (int t = 0; t < 8; ++t)
    if (j == nbr_row[t]) v = band[t];
  out[j] = v;
}

// ---------------------------------------------------------------------------
// Host launcher
// ---------------------------------------------------------------------------
extern "C" void kernel_launch(void* const* d_in, const int* in_sizes, int n_in,
                              void* d_out, int out_size, void* d_ws, size_t ws_size,
                              hipStream_t stream) {
  (void)in_sizes; (void)n_in; (void)out_size; (void)ws_size;
  // setup_inputs() dict order: xs, nbrs, k, m, w0,b0, w1,b1, w2,b2, w3,b3, w4,b4
  const float* xs   = (const float*)d_in[0];
  const int*   nbrs = (const int*)  d_in[1];   // (5,8) int32
  const float* w0   = (const float*)d_in[4];
  const float* b0   = (const float*)d_in[5];
  const float* w1   = (const float*)d_in[6];
  const float* b1   = (const float*)d_in[7];
  const float* w2   = (const float*)d_in[8];
  const float* b2   = (const float*)d_in[9];
  const float* w3   = (const float*)d_in[10];
  const float* b3   = (const float*)d_in[11];
  const float* w4   = (const float*)d_in[12];
  const float* b4   = (const float*)d_in[13];

  char* ws = (char*)d_ws;
  size_t off = 0;
  auto carve = [&](size_t bytes) -> void* {
    void* p = ws + off;
    off += (bytes + 255) & ~(size_t)255;
    return p;
  };
  const size_t ACT_B = (size_t)MPAD * HID * sizeof(_Float16);
  const size_t WT_B  = (size_t)HID * HID * sizeof(_Float16);
  _Float16* actA = (_Float16*)carve(ACT_B);
  _Float16* actB = (_Float16*)carve(ACT_B);
  _Float16* wt1  = (_Float16*)carve(WT_B);
  _Float16* wt2  = (_Float16*)carve(WT_B);
  _Float16* wt3  = (_Float16*)carve(WT_B);
  float*    amg  = (float*)carve((size_t)MPAD * sizeof(float));
  float*    bufA = (float*)carve((size_t)262145 * sizeof(float));
  float*    bufB = (float*)carve((size_t)262145 * sizeof(float));

  const int cvt_blocks = (HID * HID) / 256;
  wt_convert<<<cvt_blocks, 256, 0, stream>>>(w1, wt1);
  wt_convert<<<cvt_blocks, 256, 0, stream>>>(w2, wt2);
  wt_convert<<<cvt_blocks, 256, 0, stream>>>(w3, wt3);

  layer0<<<(MPAD * HID) / 256, 256, 0, stream>>>(xs, w0, b0, actA);

  dim3 ggrid(MPAD / BM, HID / BN);   // (129, 16)
  gemm_relu<<<ggrid, 256, 0, stream>>>(actA, wt1, b1, actB);
  gemm_relu<<<ggrid, 256, 0, stream>>>(actB, wt2, b2, actA);
  gemm_relu<<<ggrid, 256, 0, stream>>>(actA, wt3, b3, actB);

  final_dot<<<(MROWS + 7) / 8, 256, 0, stream>>>(actB, w4, b4, amg);

  // Multigrid: Ah = amg[40:], 16385 points; 5 interp+scatter rounds.
  const float* Ah = amg + 40;
  int n = 16385;
  float* outs[5] = { bufA, bufB, bufA, bufB, (float*)d_out };
  for (int i = 0; i < 5; ++i) {
    int n_out = 2 * n - 1;
    interp_scatter<<<(n_out + 255) / 256, 256, 0, stream>>>(
        Ah, n, outs[i], nbrs + 8 * i, amg + 8 * (4 - i));
    Ah = outs[i];
    n = n_out;
  }
}
